// Attention_74766790689469
// MI455X (gfx1250) — matrix operand order repaired and checked by reading, not compiled, not run on hardware
//
#include <hip/hip_runtime.h>

typedef __attribute__((ext_vector_type(16))) _Float16 v16h;
typedef __attribute__((ext_vector_type(8)))  _Float16 v8h;
typedef __attribute__((ext_vector_type(8)))  float    v8f;
typedef __attribute__((ext_vector_type(4)))  float    v4f;

#define R_TOTAL 8192   // b*n
#define NSEQ    2048
#define DIM     512
#define HEADS   8
#define DHEAD   64

__device__ __forceinline__ v8f wmma32(v16h a, v16h b, v8f c) {
  return __builtin_amdgcn_wmma_f32_16x16x32_f16(false, a, false, b, (short)0, c, false, false);
}

// A-matrix fragment (16x32, fp16) from row-major [row][k] with leading dim ld (halves).
// ISA layout: lanes 0-15 -> M=lane, K={0..7,16..23}; lanes 16-31 -> M=lane-16, K={8..15,24..31}.
__device__ __forceinline__ v16h load_a_frag(const _Float16* src, int ld, int r0, int k0, int lane) {
  int hf = lane >> 4;
  int m  = lane & 15;
  const _Float16* p = src + (r0 + m) * ld + k0;
  v8h lo = *(const v8h*)(p + 8 * hf);
  v8h hi = *(const v8h*)(p + 16 + 8 * hf);
  return __builtin_shufflevector(lo, hi, 0,1,2,3,4,5,6,7,8,9,10,11,12,13,14,15);
}

// B-matrix fragment (32x16, fp16) where B[K][N] = src[n0+N][k0+K] (src row-major, ld in halves).
// ISA layout: lanes 0-15 -> K={0..7,16..23}, lanes 16-31 -> K={8..15,24..31}, N = lane%16.
__device__ __forceinline__ v16h load_bT_frag(const _Float16* src, int ld, int n0, int k0, int lane) {
  int hf = lane >> 4;
  int nn = lane & 15;
  const _Float16* p = src + (n0 + nn) * ld + k0;
  v8h lo = *(const v8h*)(p + 8 * hf);
  v8h hi = *(const v8h*)(p + 16 + 8 * hf);
  return __builtin_shufflevector(lo, hi, 0,1,2,3,4,5,6,7,8,9,10,11,12,13,14,15);
}

// ---- CDNA5 async global->LDS copy (ASYNCcnt-tracked, no VGPR staging) ------
// Generic pointers to __shared__ carry the LDS byte offset in their low 32 bits
// (flat aperture rule).
__device__ __forceinline__ void async_cp_b128(void* lds, const void* gptr) {
  asm volatile("global_load_async_to_lds_b128 %0, %1, off"
               :: "v"((unsigned int)(unsigned long long)lds), "v"(gptr)
               : "memory");
}
#define S_WAIT_ASYNCCNT(n) asm volatile("s_wait_asynccnt " #n ::: "memory")

// ---------------------------------------------------------------------------
// Fused LayerNorm + input projection:  f = LN(x) @ w_in^T  (64 rows x 512 cols per block)
// mode 0: per-head L2-normalize rows, store [h][r][d] fp16  (q, k)
// mode 1: store transposed [h*64+d][r] fp16                 (v)
// ---------------------------------------------------------------------------
__global__ __launch_bounds__(256) void ln_proj_kernel(
    const float* __restrict__ x, const float* __restrict__ gamma,
    const float* __restrict__ beta, const float* __restrict__ w_in,
    _Float16* __restrict__ dst, int mode) {
  __shared__ _Float16 An[64 * 40];
  __shared__ _Float16 Wl[512 * 40];
  __shared__ float s_mean[64];
  __shared__ float s_rstd[64];
  const int LDA = 40;

  int tid  = threadIdx.x;
  int lane = tid & 31;
  int w    = tid >> 5;        // wave 0..7 == head
  int r0   = blockIdx.x * 64;

  // ---- LayerNorm statistics: 4 lanes per row, shfl_xor reduce ----
  {
    int row = tid >> 2, part = tid & 3;
    const float* xr = x + (size_t)(r0 + row) * DIM + part * 128;
    float s = 0.f, sq = 0.f;
#pragma unroll
    for (int i = 0; i < 32; ++i) {
      v4f v = *(const v4f*)(xr + i * 4);
#pragma unroll
      for (int j = 0; j < 4; ++j) { s += v[j]; sq += v[j] * v[j]; }
    }
    s  += __shfl_xor(s, 1);  s  += __shfl_xor(s, 2);
    sq += __shfl_xor(sq, 1); sq += __shfl_xor(sq, 2);
    if (part == 0) {
      float mu  = s * (1.f / 512.f);
      float var = sq * (1.f / 512.f) - mu * mu;
      s_mean[row] = mu;
      s_rstd[row] = rsqrtf(var + 1e-5f);
    }
  }
  __syncthreads();

  v8f zero = {0.f,0.f,0.f,0.f,0.f,0.f,0.f,0.f};
  v8f acc[4][4];
#pragma unroll
  for (int i = 0; i < 4; ++i)
#pragma unroll
    for (int j = 0; j < 4; ++j) acc[i][j] = zero;

  for (int kc = 0; kc < 16; ++kc) {
    int k0 = kc * 32;
    // stage normalized A chunk (64x32 fp16)
    {
      int row = tid >> 2, c = (tid & 3) * 8;
      float mu = s_mean[row], rs = s_rstd[row];
      const float* xr = x + (size_t)(r0 + row) * DIM + k0 + c;
      v4f v0 = *(const v4f*)(xr);
      v4f v1 = *(const v4f*)(xr + 4);
      v4f g0 = *(const v4f*)(gamma + k0 + c);
      v4f g1 = *(const v4f*)(gamma + k0 + c + 4);
      v4f b0 = *(const v4f*)(beta + k0 + c);
      v4f b1 = *(const v4f*)(beta + k0 + c + 4);
      _Float16* ap = An + row * LDA + c;
#pragma unroll
      for (int i = 0; i < 4; ++i) {
        ap[i]     = (_Float16)((v0[i] - mu) * rs * g0[i] + b0[i]);
        ap[4 + i] = (_Float16)((v1[i] - mu) * rs * g1[i] + b1[i]);
      }
    }
    // stage weight chunk Wl[j][kk] = w_in[j][k0+kk]  (512x32 fp16)
    {
      int c = (tid & 3) * 8, jb = tid >> 2;
#pragma unroll
      for (int rep = 0; rep < 8; ++rep) {
        int j = rep * 64 + jb;
        const float* wr = w_in + (size_t)j * DIM + k0 + c;
        v4f a = *(const v4f*)(wr);
        v4f b = *(const v4f*)(wr + 4);
        _Float16* wp = Wl + j * LDA + c;
#pragma unroll
        for (int i = 0; i < 4; ++i) { wp[i] = (_Float16)a[i]; wp[4 + i] = (_Float16)b[i]; }
      }
    }
    __syncthreads();

    v16h af[4], bf[4];
#pragma unroll
    for (int mt = 0; mt < 4; ++mt) af[mt] = load_a_frag(An, LDA, mt * 16, 0, lane);
#pragma unroll
    for (int ct = 0; ct < 4; ++ct) bf[ct] = load_bT_frag(Wl, LDA, w * 64 + ct * 16, 0, lane);
#pragma unroll
    for (int mt = 0; mt < 4; ++mt)
#pragma unroll
      for (int ct = 0; ct < 4; ++ct) acc[mt][ct] = wmma32(af[mt], bf[ct], acc[mt][ct]);
    __syncthreads();
  }

  // ---- epilogue: wave w holds head w's 64 columns for 64 rows ----
  int hf = lane >> 4, nn = lane & 15;
  if (mode == 0) {
#pragma unroll
    for (int mt = 0; mt < 4; ++mt) {
#pragma unroll
      for (int g = 0; g < 8; ++g) {
        float sq = 0.f;
#pragma unroll
        for (int ct = 0; ct < 4; ++ct) { float t = acc[mt][ct][g]; sq += t * t; }
        sq += __shfl_xor(sq, 1); sq += __shfl_xor(sq, 2);
        sq += __shfl_xor(sq, 4); sq += __shfl_xor(sq, 8);
        float rn = rsqrtf(sq + 1e-20f);
        int rgl = r0 + mt * 16 + g + 8 * hf;
        _Float16* d = dst + ((size_t)w * R_TOTAL + rgl) * DHEAD;
#pragma unroll
        for (int ct = 0; ct < 4; ++ct) d[ct * 16 + nn] = (_Float16)(acc[mt][ct][g] * rn);
      }
    }
  } else {
#pragma unroll
    for (int mt = 0; mt < 4; ++mt)
#pragma unroll
      for (int g = 0; g < 8; ++g) {
        int rgl = r0 + mt * 16 + g + 8 * hf;
#pragma unroll
        for (int ct = 0; ct < 4; ++ct)
          dst[((size_t)(w * 64 + ct * 16 + nn)) * R_TOTAL + rgl] = (_Float16)acc[mt][ct][g];
      }
  }
}

// ---------------------------------------------------------------------------
// Cosine attention (no softmax): O = (q̂ k̂ᵀ) v   per (head, batch)
// block = 64 query rows; streams 2048 keys in 64-wide tiles through LDS using
// double-buffered GLOBAL_LOAD_ASYNC_TO_LDS_B128 (ASYNCcnt-synchronized).
// ---------------------------------------------------------------------------
__global__ __launch_bounds__(256) void attn_kernel(
    const _Float16* __restrict__ qn,   // [8][8192][64]
    const _Float16* __restrict__ kn,   // [8][8192][64]
    const _Float16* __restrict__ vt,   // [512][8192]  ([h*64+d][r])
    float* __restrict__ outa) {        // [8192][512], col = h*64+d
  __shared__ _Float16 Ql[64 * 72];
  __shared__ _Float16 Kl[2][64 * 72];
  __shared__ _Float16 Vl[2][64 * 72];
  __shared__ _Float16 Sl[64 * 72];
  const int LD = 72;

  int tid = threadIdx.x, lane = tid & 31, w = tid >> 5;
  int qi = blockIdx.x;            // query tile 0..31
  int hb = blockIdx.y;            // 0..31
  int h  = hb >> 2, bb = hb & 3;
  int rb = bb * NSEQ;
  int q0 = rb + qi * 64;

  int row = tid >> 2;             // 0..63
  int c8  = (tid & 3) * 16;       // 0,16,32,48 (halves)

  const _Float16* kbase = kn + ((size_t)h * R_TOTAL + rb) * DHEAD;
  const _Float16* vbase = vt + (size_t)h * 64 * R_TOTAL + rb;

  // async-load resident Q tile (2 instr/wave), then tile 0 of K/V (4 instr/wave)
  {
    const _Float16* qs = qn + ((size_t)h * R_TOTAL + q0) * DHEAD + (size_t)row * DHEAD + c8;
    async_cp_b128(Ql + row * LD + c8,     qs);
    async_cp_b128(Ql + row * LD + c8 + 8, qs + 8);
  }
  {
    const _Float16* ks = kbase + (size_t)row * DHEAD + c8;
    const _Float16* vs = vbase + (size_t)row * R_TOTAL + c8;
    async_cp_b128(Kl[0] + row * LD + c8,     ks);
    async_cp_b128(Kl[0] + row * LD + c8 + 8, ks + 8);
    async_cp_b128(Vl[0] + row * LD + c8,     vs);
    async_cp_b128(Vl[0] + row * LD + c8 + 8, vs + 8);
  }
  S_WAIT_ASYNCCNT(4);   // Q tile complete (async loads finish in order)
  __syncthreads();

  int mq = w & 3;              // row tile of this wave
  int cs = (w >> 2) * 32;      // 32-col slab of this wave

  v16h aq0 = load_a_frag(Ql, LD, mq * 16, 0, lane);
  v16h aq1 = load_a_frag(Ql, LD, mq * 16, 32, lane);

  v8f o0 = {0.f,0.f,0.f,0.f,0.f,0.f,0.f,0.f};
  v8f o1 = o0;

  for (int it = 0; it < NSEQ / 64; ++it) {
    int cur = it & 1;
    // prefetch next K/V tile into the alternate buffer while computing
    if (it + 1 < NSEQ / 64) {
      int m1 = (it + 1) * 64;
      const _Float16* ks = kbase + (size_t)(m1 + row) * DHEAD + c8;
      const _Float16* vs = vbase + (size_t)row * R_TOTAL + m1 + c8;
      _Float16* kd = Kl[cur ^ 1] + row * LD + c8;
      _Float16* vd = Vl[cur ^ 1] + row * LD + c8;
      async_cp_b128(kd,     ks);
      async_cp_b128(kd + 8, ks + 8);
      async_cp_b128(vd,     vs);
      async_cp_b128(vd + 8, vs + 8);
      S_WAIT_ASYNCCNT(4);    // tile `it` resident; only the 4 prefetches pending
    } else {
      S_WAIT_ASYNCCNT(0);
    }
    __syncthreads();

    const _Float16* Kc = Kl[cur];
    const _Float16* Vc = Vl[cur];

    // S = Q K^T  (this wave: rows mq*16.., cols cs..cs+31)
    int hf = lane >> 4, nn = lane & 15;
#pragma unroll
    for (int ci = 0; ci < 2; ++ci) {
      int ct = cs + ci * 16;
      v8f s = {0.f,0.f,0.f,0.f,0.f,0.f,0.f,0.f};
      s = wmma32(aq0, load_bT_frag(Kc, LD, ct, 0, lane), s);
      s = wmma32(aq1, load_bT_frag(Kc, LD, ct, 32, lane), s);
#pragma unroll
      for (int g = 0; g < 8; ++g)
        Sl[(mq * 16 + g + 8 * hf) * LD + ct + nn] = (_Float16)s[g];
    }
    __syncthreads();

    // O += S V
#pragma unroll
    for (int kc = 0; kc < 2; ++kc) {
      v16h as  = load_a_frag(Sl, LD, mq * 16, kc * 32, lane);
      v16h bv0 = load_bT_frag(Vc, LD, cs, kc * 32, lane);
      v16h bv1 = load_bT_frag(Vc, LD, cs + 16, kc * 32, lane);
      o0 = wmma32(as, bv0, o0);
      o1 = wmma32(as, bv1, o1);
    }
    __syncthreads();   // buffer `cur` free for the prefetch two iterations out
  }

  int hf = lane >> 4, nn = lane & 15;
#pragma unroll
  for (int g = 0; g < 8; ++g) {
    int rgl = q0 + mq * 16 + g + 8 * hf;
    float* d = outa + (size_t)rgl * DIM + h * 64 + cs;
    d[nn]      = o0[g];
    d[16 + nn] = o1[g];
  }
}

// ---------------------------------------------------------------------------
// Output projection: out = outa @ w_out^T + b_out   (fp32 in/out, fp16 WMMA)
// ---------------------------------------------------------------------------
__global__ __launch_bounds__(256) void proj_out_kernel(
    const float* __restrict__ xin, const float* __restrict__ w_out,
    const float* __restrict__ b_out, float* __restrict__ out) {
  __shared__ _Float16 An[64 * 40];
  __shared__ _Float16 Wl[512 * 40];
  const int LDA = 40;

  int tid = threadIdx.x, lane = tid & 31, w = tid >> 5;
  int r0 = blockIdx.x * 64;

  v8f zero = {0.f,0.f,0.f,0.f,0.f,0.f,0.f,0.f};
  v8f acc[4][4];
#pragma unroll
  for (int i = 0; i < 4; ++i)
#pragma unroll
    for (int j = 0; j < 4; ++j) acc[i][j] = zero;

  for (int kc = 0; kc < 16; ++kc) {
    int k0 = kc * 32;
    {
      int row = tid >> 2, c = (tid & 3) * 8;
      const float* xr = xin + (size_t)(r0 + row) * DIM + k0 + c;
      v4f v0 = *(const v4f*)(xr);
      v4f v1 = *(const v4f*)(xr + 4);
      _Float16* ap = An + row * LDA + c;
#pragma unroll
      for (int i = 0; i < 4; ++i) { ap[i] = (_Float16)v0[i]; ap[4 + i] = (_Float16)v1[i]; }
    }
    {
      int c = (tid & 3) * 8, jb = tid >> 2;
#pragma unroll
      for (int rep = 0; rep < 8; ++rep) {
        int j = rep * 64 + jb;
        const float* wr = w_out + (size_t)j * DIM + k0 + c;
        v4f a = *(const v4f*)(wr);
        v4f b = *(const v4f*)(wr + 4);
        _Float16* wp = Wl + j * LDA + c;
#pragma unroll
        for (int i = 0; i < 4; ++i) { wp[i] = (_Float16)a[i]; wp[4 + i] = (_Float16)b[i]; }
      }
    }
    __syncthreads();

    v16h af[4], bf[4];
#pragma unroll
    for (int mt = 0; mt < 4; ++mt) af[mt] = load_a_frag(An, LDA, mt * 16, 0, lane);
#pragma unroll
    for (int ct = 0; ct < 4; ++ct) bf[ct] = load_bT_frag(Wl, LDA, w * 64 + ct * 16, 0, lane);
#pragma unroll
    for (int mt = 0; mt < 4; ++mt)
#pragma unroll
      for (int ct = 0; ct < 4; ++ct) acc[mt][ct] = wmma32(af[mt], bf[ct], acc[mt][ct]);
    __syncthreads();
  }

  int hf = lane >> 4, nn = lane & 15;
#pragma unroll
  for (int mt = 0; mt < 4; ++mt)
#pragma unroll
    for (int g = 0; g < 8; ++g) {
      int rgl = r0 + mt * 16 + g + 8 * hf;
#pragma unroll
      for (int ct = 0; ct < 4; ++ct) {
        int col = w * 64 + ct * 16 + nn;
        out[(size_t)rgl * DIM + col] = acc[mt][ct][g] + b_out[col];
      }
    }
}

// ---------------------------------------------------------------------------
extern "C" void kernel_launch(void* const* d_in, const int* in_sizes, int n_in,
                              void* d_out, int out_size, void* d_ws, size_t ws_size,
                              hipStream_t stream) {
  const float* q        = (const float*)d_in[0];
  const float* k        = (const float*)d_in[1];
  const float* v        = (const float*)d_in[2];
  const float* ln_gamma = (const float*)d_in[3];
  const float* ln_beta  = (const float*)d_in[4];
  const float* w_in     = (const float*)d_in[5];
  const float* w_out    = (const float*)d_in[6];
  const float* b_out    = (const float*)d_in[7];
  float* out = (float*)d_out;

  const size_t half_tensor = (size_t)HEADS * R_TOTAL * DHEAD * sizeof(_Float16); // 8 MB
  char* ws = (char*)d_ws;
  _Float16* qn  = (_Float16*)(ws);
  _Float16* kn  = (_Float16*)(ws + half_tensor);
  _Float16* vt  = (_Float16*)(ws + 2 * half_tensor);
  float*    outa = (float*)(ws + 3 * half_tensor);                               // 16 MB

  ln_proj_kernel<<<128, 256, 0, stream>>>(q, ln_gamma, ln_beta, w_in, qn, 0);
  ln_proj_kernel<<<128, 256, 0, stream>>>(k, ln_gamma, ln_beta, w_in, kn, 0);
  ln_proj_kernel<<<128, 256, 0, stream>>>(v, ln_gamma, ln_beta, w_in, vt, 1);
  attn_kernel<<<dim3(32, 32), 256, 0, stream>>>(qn, kn, vt, outa);
  proj_out_kernel<<<128, 256, 0, stream>>>(outa, w_out, b_out, out);
}
